// CrossModalGraphLayer_18270790877214
// MI455X (gfx1250) — matrix operand order repaired and checked by reading, not compiled, not run on hardware
//
#include <hip/hip_runtime.h>

// ---------------------------------------------------------------------------
// CrossModalGraphLayer for MI455X (gfx1250, wave32)
//   proj  = input @ W1^T                        (bf16 WMMA, f32 accum)
//   neigh = segment_sum(val * proj[col], row)   (f32 atomics, L2-resident)
//   out   = leaky_relu([x+n, x*n] @ W2^T)       (bf16 WMMA, f32 accum)
// ---------------------------------------------------------------------------

#define NN      100000      // nodes
#define DIM     128         // feature dim
#define NE      1600000     // edges
#define SLOPE   0.01f

typedef __attribute__((ext_vector_type(16))) __bf16 v16bf;
typedef __attribute__((ext_vector_type(8)))  float  v8f;

// ---- workspace layout (bytes) ----
static constexpr size_t PROJ_BYTES = (size_t)NN * DIM * sizeof(float);  // 51.2 MB
static constexpr size_t PROJ_OFF   = 0;
static constexpr size_t NEIGH_OFF  = PROJ_OFF + PROJ_BYTES;             // 51.2 MB
static constexpr size_t W1B_OFF    = NEIGH_OFF + PROJ_BYTES;            // +32 KB (bf16)
static constexpr size_t W2B_OFF    = W1B_OFF + (size_t)DIM * DIM * 2;   // +64 KB (bf16)

// ---------------------------------------------------------------------------
// Fragment loaders from LDS, per CDNA5 ISA 7.12.2 (wave32 layouts).
// A (16x32 bf16, MxK): lane half h=lane>>4, row M=lane&15.
//   elements 0..7  -> K = kBase + h*8 + 0..7
//   elements 8..15 -> K = kBase + 16 + h*8 + 0..7
// B (32x16 bf16, KxN): col N = lane&15; same K order as A:
//   elements 0..7 -> K = kBase + h*8 + 0..7, 8..15 -> K = kBase + 16 + h*8 + 0..7.
// All offsets 16B-aligned -> ds_load_b128.
// ---------------------------------------------------------------------------
__device__ __forceinline__ v16bf load_a_frag(const __bf16* base, int lane,
                                             int kBase, int ld) {
  const int half = (lane >> 4) & 1;
  const int m    = lane & 15;
  const __bf16* p = base + m * ld + kBase + half * 8;
  v16bf f;
#pragma unroll
  for (int i = 0; i < 8; ++i) { f[i] = p[i]; f[8 + i] = p[16 + i]; }
  return f;
}

__device__ __forceinline__ v16bf load_b_frag(const __bf16* wbase, int lane,
                                             int n0, int kBase, int ld) {
  const int half = (lane >> 4) & 1;
  const int n    = lane & 15;
  const __bf16* p = wbase + (size_t)(n0 + n) * ld + kBase + half * 8;
  v16bf f;
#pragma unroll
  for (int i = 0; i < 8; ++i) { f[i] = p[i]; f[8 + i] = p[16 + i]; }
  return f;
}

// ---------------------------------------------------------------------------
// Kernel 1: convert W1 [128x128] and W2 [128x256] (row-major [out][in]) to
// bf16 in workspace. [out][in] == [N][K] layout needed by B fragments.
// ---------------------------------------------------------------------------
__global__ __launch_bounds__(256) void wconv_kernel(
    const float* __restrict__ W1, const float* __restrict__ W2,
    char* __restrict__ ws) {
  __bf16* w1b = (__bf16*)(ws + W1B_OFF);
  __bf16* w2b = (__bf16*)(ws + W2B_OFF);
  int i = blockIdx.x * 256 + threadIdx.x;
  if (i < DIM * DIM)     w1b[i] = (__bf16)W1[i];
  if (i < DIM * 2 * DIM) w2b[i] = (__bf16)W2[i];
}

// ---------------------------------------------------------------------------
// Kernel 2: zero the neigh accumulator (harness poisons ws with 0xAA).
// ---------------------------------------------------------------------------
__global__ __launch_bounds__(256) void zero_kernel(char* __restrict__ ws) {
  float4* p = (float4*)(ws + NEIGH_OFF);
  int i = blockIdx.x * 256 + threadIdx.x;
  if (i < NN * DIM / 4) p[i] = make_float4(0.f, 0.f, 0.f, 0.f);
}

// ---------------------------------------------------------------------------
// Kernel 3: proj = input @ W1^T via V_WMMA_F32_16X16X32_BF16.
// 256 threads = 8 waves; 128-row tile; each wave owns 16 rows x 128 cols.
// Dynamic LDS: A tile [128][136] bf16 + W1 [128][136] bf16 (= 68 KB).
// Tail handling is block-level: only the last of 782 blocks is partial.
// ---------------------------------------------------------------------------
__global__ __launch_bounds__(256, 2) void gemm1_kernel(
    const float* __restrict__ input, char* __restrict__ ws) {
  constexpr int LDA = DIM + 8;  // 136: +16B pad, keeps 16B alignment, skews banks
  extern __shared__ char smem[];
  __bf16* As = (__bf16*)smem;          // [128][LDA]
  __bf16* Wsh = As + 128 * LDA;        // [128][LDA]
  const __bf16* w1b = (const __bf16*)(ws + W1B_OFF);
  float* proj = (float*)(ws + PROJ_OFF);

  const int row0 = blockIdx.x * 128;
  const int tid  = threadIdx.x;
  const bool full = (row0 + 128 <= NN);   // block-uniform

  // stage input tile, f32 -> bf16 (float4 granularity, 16 iters/thread)
  if (full) {
#pragma unroll 4
    for (int idx = tid; idx < 128 * (DIM / 4); idx += 256) {
      int m  = idx / (DIM / 4);
      int k4 = (idx % (DIM / 4)) * 4;
      float4 v = *(const float4*)(input + (size_t)(row0 + m) * DIM + k4);
      __bf16* d = As + m * LDA + k4;
      d[0] = (__bf16)v.x; d[1] = (__bf16)v.y; d[2] = (__bf16)v.z; d[3] = (__bf16)v.w;
    }
  } else {
    for (int idx = tid; idx < 128 * (DIM / 4); idx += 256) {
      int m  = idx / (DIM / 4);
      int k4 = (idx % (DIM / 4)) * 4;
      float4 v = make_float4(0.f, 0.f, 0.f, 0.f);
      if (row0 + m < NN) v = *(const float4*)(input + (size_t)(row0 + m) * DIM + k4);
      __bf16* d = As + m * LDA + k4;
      d[0] = (__bf16)v.x; d[1] = (__bf16)v.y; d[2] = (__bf16)v.z; d[3] = (__bf16)v.w;
    }
  }
  // stage W1 bf16 (hot in L2 after first block) as dword copies
#pragma unroll 4
  for (int idx = tid; idx < 128 * (DIM / 2); idx += 256) {
    int n  = idx / (DIM / 2);
    int k2 = (idx % (DIM / 2)) * 2;
    *(unsigned int*)(Wsh + n * LDA + k2) = ((const unsigned int*)w1b)[idx];
  }
  __syncthreads();

  const int wave = tid >> 5, lane = tid & 31;
  const int m0 = wave * 16;
  const __bf16* abase = As + m0 * LDA;

  v16bf afr[4];
#pragma unroll
  for (int kk = 0; kk < 4; ++kk) afr[kk] = load_a_frag(abase, lane, kk * 32, LDA);

  // single 64-bit base per wave; all stores use constant immediate offsets
  const int half = lane >> 4, n = lane & 15;
  const int gm = row0 + m0 + half * 8;           // first output row for this lane
  float* const obase = proj + (size_t)gm * DIM + n;

#pragma unroll
  for (int n0 = 0; n0 < DIM; n0 += 16) {
    v8f acc = {0.f, 0.f, 0.f, 0.f, 0.f, 0.f, 0.f, 0.f};
#pragma unroll
    for (int kk = 0; kk < 4; ++kk) {
      v16bf b = load_b_frag(Wsh, lane, n0, kk * 32, LDA);
      acc = __builtin_amdgcn_wmma_f32_16x16x32_bf16(
          false, afr[kk], false, b, (short)0, acc, false, false);
    }
    // C layout: VGPR r -> (M = r + 8*half, N = lane&15)
    if (full) {
#pragma unroll
      for (int r = 0; r < 8; ++r) obase[r * DIM + n0] = acc[r];
    } else {
#pragma unroll
      for (int r = 0; r < 8; ++r)
        if (gm + r < NN) obase[r * DIM + n0] = acc[r];
    }
  }
}

// ---------------------------------------------------------------------------
// Kernel 4: edge scatter. One wave32 per edge; lane handles 4 contiguous
// features (128 = 32*4): global_load_b128 gather from proj (L2-resident) +
// 4x global_atomic_add_f32 into neigh (L2-resident).
// ---------------------------------------------------------------------------
__global__ __launch_bounds__(256) void edge_scatter_kernel(
    const int* __restrict__ erow, const int* __restrict__ ecol,
    const float* __restrict__ eval, char* __restrict__ ws) {
  const float* proj = (const float*)(ws + PROJ_OFF);
  float* neigh = (float*)(ws + NEIGH_OFF);
  unsigned gt = blockIdx.x * 256u + threadIdx.x;
  int e = (int)(gt >> 5);
  if (e >= NE) return;
  int lane = threadIdx.x & 31;
  int r = erow[e];
  int c = ecol[e];
  float v = eval[e];
  float4 p = *((const float4*)(proj + (size_t)c * DIM) + lane);
  float* dst = neigh + (size_t)r * DIM + lane * 4;
  atomicAdd(dst + 0, v * p.x);
  atomicAdd(dst + 1, v * p.y);
  atomicAdd(dst + 2, v * p.z);
  atomicAdd(dst + 3, v * p.w);
}

// ---------------------------------------------------------------------------
// Kernel 5: out = leaky_relu([x+n, x*n] @ W2^T).
// H tile [128][264] bf16 + W2 [128][264] bf16 in dynamic LDS (= 132 KB,
// legal on the 320 KB WGP). 8 K-steps of 32 over K=256.
// ---------------------------------------------------------------------------
__global__ __launch_bounds__(256, 1) void gemm2_kernel(
    const float* __restrict__ input, char* __restrict__ ws,
    float* __restrict__ out) {
  constexpr int K2  = 2 * DIM;   // 256
  constexpr int LDH = K2 + 8;    // 264
  extern __shared__ char smem[];
  __bf16* Hs  = (__bf16*)smem;          // [128][LDH]
  __bf16* Wsh = Hs + 128 * LDH;         // [128][LDH]
  const float* neigh = (const float*)(ws + NEIGH_OFF);
  const __bf16* w2b  = (const __bf16*)(ws + W2B_OFF);

  const int row0 = blockIdx.x * 128;
  const int tid  = threadIdx.x;
  const bool full = (row0 + 128 <= NN);   // block-uniform

  // build h = [x + n | x * n] in bf16
  if (full) {
#pragma unroll 4
    for (int idx = tid; idx < 128 * (DIM / 4); idx += 256) {
      int m  = idx / (DIM / 4);
      int k4 = (idx % (DIM / 4)) * 4;
      float4 a = *(const float4*)(input + (size_t)(row0 + m) * DIM + k4);
      float4 b = *(const float4*)(neigh + (size_t)(row0 + m) * DIM + k4);
      __bf16* d0 = Hs + m * LDH + k4;        // x + n
      __bf16* d1 = d0 + DIM;                 // x * n
      d0[0] = (__bf16)(a.x + b.x); d0[1] = (__bf16)(a.y + b.y);
      d0[2] = (__bf16)(a.z + b.z); d0[3] = (__bf16)(a.w + b.w);
      d1[0] = (__bf16)(a.x * b.x); d1[1] = (__bf16)(a.y * b.y);
      d1[2] = (__bf16)(a.z * b.z); d1[3] = (__bf16)(a.w * b.w);
    }
  } else {
    for (int idx = tid; idx < 128 * (DIM / 4); idx += 256) {
      int m  = idx / (DIM / 4);
      int k4 = (idx % (DIM / 4)) * 4;
      float4 a = make_float4(0.f, 0.f, 0.f, 0.f);
      float4 b = make_float4(0.f, 0.f, 0.f, 0.f);
      if (row0 + m < NN) {
        a = *(const float4*)(input + (size_t)(row0 + m) * DIM + k4);
        b = *(const float4*)(neigh + (size_t)(row0 + m) * DIM + k4);
      }
      __bf16* d0 = Hs + m * LDH + k4;
      __bf16* d1 = d0 + DIM;
      d0[0] = (__bf16)(a.x + b.x); d0[1] = (__bf16)(a.y + b.y);
      d0[2] = (__bf16)(a.z + b.z); d0[3] = (__bf16)(a.w + b.w);
      d1[0] = (__bf16)(a.x * b.x); d1[1] = (__bf16)(a.y * b.y);
      d1[2] = (__bf16)(a.z * b.z); d1[3] = (__bf16)(a.w * b.w);
    }
  }
  // stage W2 bf16
#pragma unroll 4
  for (int idx = tid; idx < 128 * (K2 / 2); idx += 256) {
    int n  = idx / (K2 / 2);
    int k2 = (idx % (K2 / 2)) * 2;
    *(unsigned int*)(Wsh + n * LDH + k2) = ((const unsigned int*)w2b)[idx];
  }
  __syncthreads();

  const int wave = tid >> 5, lane = tid & 31;
  const int m0 = wave * 16;
  const __bf16* abase = Hs + m0 * LDH;

  v16bf afr[8];
#pragma unroll
  for (int kk = 0; kk < 8; ++kk) afr[kk] = load_a_frag(abase, lane, kk * 32, LDH);

  const int half = lane >> 4, n = lane & 15;
  const int gm = row0 + m0 + half * 8;
  float* const obase = out + (size_t)gm * DIM + n;

#pragma unroll
  for (int n0 = 0; n0 < DIM; n0 += 16) {
    v8f acc = {0.f, 0.f, 0.f, 0.f, 0.f, 0.f, 0.f, 0.f};
#pragma unroll
    for (int kk = 0; kk < 8; ++kk) {
      v16bf b = load_b_frag(Wsh, lane, n0, kk * 32, LDH);
      acc = __builtin_amdgcn_wmma_f32_16x16x32_bf16(
          false, afr[kk], false, b, (short)0, acc, false, false);
    }
    if (full) {
#pragma unroll
      for (int r = 0; r < 8; ++r) {
        float x = acc[r];
        obase[r * DIM + n0] = (x > 0.f) ? x : SLOPE * x;
      }
    } else {
#pragma unroll
      for (int r = 0; r < 8; ++r) {
        if (gm + r < NN) {
          float x = acc[r];
          obase[r * DIM + n0] = (x > 0.f) ? x : SLOPE * x;
        }
      }
    }
  }
}

// ---------------------------------------------------------------------------
extern "C" void kernel_launch(void* const* d_in, const int* in_sizes, int n_in,
                              void* d_out, int out_size, void* d_ws,
                              size_t ws_size, hipStream_t stream) {
  (void)in_sizes; (void)n_in; (void)out_size; (void)ws_size;

  const float* input = (const float*)d_in[0];
  const int*   erow  = (const int*)d_in[1];
  const int*   ecol  = (const int*)d_in[2];
  const float* eval  = (const float*)d_in[3];
  const float* W1    = (const float*)d_in[4];
  const float* W2    = (const float*)d_in[5];
  float* out = (float*)d_out;
  char*  ws  = (char*)d_ws;

  const int rowBlocks = (NN + 127) / 128;                     // 782
  const size_t lds1 = (size_t)(128 * (DIM + 8)) * 2 * 2;      // 68 KB
  const size_t lds2 = (size_t)(128 * (2 * DIM + 8)) * 2 * 2;  // 132 KB

  wconv_kernel<<<(DIM * 2 * DIM + 255) / 256, 256, 0, stream>>>(W1, W2, ws);
  zero_kernel<<<(NN * DIM / 4 + 255) / 256, 256, 0, stream>>>(ws);
  gemm1_kernel<<<rowBlocks, 256, lds1, stream>>>(input, ws);
  edge_scatter_kernel<<<(NE * 32 + 255) / 256, 256, 0, stream>>>(erow, ecol, eval, ws);
  gemm2_kernel<<<rowBlocks, 256, lds2, stream>>>(input, ws, out);
}
